// SAGE_41970420418158
// MI455X (gfx1250) — matrix operand order repaired and checked by reading, not compiled, not run on hardware
//
#include <hip/hip_runtime.h>
#include <hip/hip_bf16.h>
#include <stdint.h>

typedef __attribute__((ext_vector_type(16))) _Float16 v16h;
typedef __attribute__((ext_vector_type(4)))  _Float16 v4h;
typedef __attribute__((ext_vector_type(8)))  float    v8f;

#define DHID 128   // hidden width (all GEMMs are K=N=128)
#define NGRAPH 512

// ---------------------------------------------------------------- utilities
__global__ void k_fill_f32(float* p, float v, int n) {
    int i = blockIdx.x * blockDim.x + threadIdx.x;
    if (i < n) p[i] = v;
}

__global__ void k_zero_f32(float* p, int n) {
    int i = blockIdx.x * blockDim.x + threadIdx.x;
    if (i < n) p[i] = 0.0f;
}

// deg += 1 for every incoming edge (self loop already counted via init=1)
__global__ void k_deg(const int* dst, float* deg, int E) {
    int i = blockIdx.x * blockDim.x + threadIdx.x;
    if (i < E) atomicAdd(&deg[dst[i]], 1.0f);
}

__global__ void k_rsqrt_inplace(float* p, int n) {
    int i = blockIdx.x * blockDim.x + threadIdx.x;
    if (i < n) p[i] = rsqrtf(p[i]);   // deg >= 1 always (self loops)
}

// 4 channels / thread: b128 load -> b64 store
__global__ void k_f32_to_f16_v4(const float* __restrict__ in, _Float16* __restrict__ out, int n4) {
    int i = blockIdx.x * blockDim.x + threadIdx.x;
    if (i >= n4) return;
    float4 v = ((const float4*)in)[i];
    v4h h = { (_Float16)v.x, (_Float16)v.y, (_Float16)v.z, (_Float16)v.w };
    ((v4h*)out)[i] = h;
}

// ------------------------------------------------- weight repack (B-fragment)
// B packing used here (32x16 f16 chunk), contiguous k order matching A: grp = lane>>4, n = lane&15,
// half i (0..15) holds element (k = 32*t4 + 16*grp + i, col = 16*w + n).
// Pack so each lane's 16 halves are one contiguous 32-byte v16h.
__global__ void k_repack_w(const float* W, _Float16* Bp) {
    int idx = blockIdx.x * blockDim.x + threadIdx.x;   // 8*4*32*16 = 16384
    if (idx >= 8 * 4 * 32 * 16) return;
    int i    = idx & 15;
    int lane = (idx >> 4) & 31;
    int t4   = (idx >> 9) & 3;
    int w    = idx >> 11;
    int grp  = lane >> 4;
    int n    = lane & 15;
    int k    = 32 * t4 + 16 * grp + i;
    Bp[idx] = (_Float16)W[k * DHID + 16 * w + n];
}

// ------------------------------------------------------------- WMMA GEMM
// C[M x 128] = A_f16[M x 128] @ B (prepacked fragments), f32 accumulate.
// Block: 256 threads = 8 waves; block owns a 64-row M tile staged in LDS.
// Wave w owns the 16-col N-tile [16w,16w+16) and loops 4 M-subtiles, keeping
// its 4 B fragments in registers (16 WMMAs per wave).
// FUSE: also write AGG = C * dinv[row]^2 (GCN self-loop term).
template <bool FUSE>
__global__ __launch_bounds__(256)
void k_gemm_wmma(const _Float16* __restrict__ A, const _Float16* __restrict__ Bp,
                 float* __restrict__ C, const float* __restrict__ dinv,
                 float* __restrict__ AGG, int M) {
    // 136-half row stride: 272 B = 16 B aligned; bank advance 4/row -> conflict free
    __shared__ _Float16 As[64 * 136];

    int row0 = blockIdx.x * 64;
    int t    = threadIdx.x;

    // cooperative A-tile load: 64 rows x 128 halves = 1024 8-half chunks, 4/thread
    #pragma unroll
    for (int it = 0; it < 4; ++it) {
        int chunk = t + 256 * it;
        int r  = chunk >> 4;       // 0..63
        int cb = chunk & 15;       // chunk of 8 halves
        uint4 av = make_uint4(0u, 0u, 0u, 0u);
        int grow = row0 + r;
        if (grow < M)
            av = *(const uint4*)(A + (size_t)grow * DHID + cb * 8);
        *(uint4*)&As[r * 136 + cb * 8] = av;
    }
    __syncthreads();

    int wave = t >> 5;         // N tile 0..7
    int lane = t & 31;
    int m    = lane & 15;
    int grp  = lane >> 4;
    int col  = wave * 16 + m;

    // B fragments: resident across all 4 M-subtiles
    v16h b0 = *(const v16h*)(Bp + (size_t)((wave * 4 + 0) * 32 + lane) * 16);
    v16h b1 = *(const v16h*)(Bp + (size_t)((wave * 4 + 1) * 32 + lane) * 16);
    v16h b2 = *(const v16h*)(Bp + (size_t)((wave * 4 + 2) * 32 + lane) * 16);
    v16h b3 = *(const v16h*)(Bp + (size_t)((wave * 4 + 3) * 32 + lane) * 16);

    #pragma unroll
    for (int msub = 0; msub < 4; ++msub) {
        int r0 = row0 + msub * 16;
        if (r0 >= M) break;

        // A fragments from LDS: halves 0-15 -> K = 32*t4 + 16*grp + 0..15
        const _Float16* arow = &As[(msub * 16 + m) * 136];
        union { v16h v; uint4 q[2]; } a0, a1, a2, a3;
        a0.q[0] = *(const uint4*)(arow + 0   + 16 * grp);
        a0.q[1] = *(const uint4*)(arow + 8   + 16 * grp);
        a1.q[0] = *(const uint4*)(arow + 32  + 16 * grp);
        a1.q[1] = *(const uint4*)(arow + 40  + 16 * grp);
        a2.q[0] = *(const uint4*)(arow + 64  + 16 * grp);
        a2.q[1] = *(const uint4*)(arow + 72  + 16 * grp);
        a3.q[0] = *(const uint4*)(arow + 96  + 16 * grp);
        a3.q[1] = *(const uint4*)(arow + 104 + 16 * grp);

        v8f acc = {};
        acc = __builtin_amdgcn_wmma_f32_16x16x32_f16(false, a0.v, false, b0, (short)0, acc, false, false);
        acc = __builtin_amdgcn_wmma_f32_16x16x32_f16(false, a1.v, false, b1, (short)0, acc, false, false);
        acc = __builtin_amdgcn_wmma_f32_16x16x32_f16(false, a2.v, false, b2, (short)0, acc, false, false);
        acc = __builtin_amdgcn_wmma_f32_16x16x32_f16(false, a3.v, false, b3, (short)0, acc, false, false);

        // C/D layout: VGPR p -> row (r0 + 8*grp) + p, col = wave*16 + (lane&15)
        int rbase = r0 + 8 * grp;
        float*       cb = C + (size_t)rbase * DHID + col;
        const float* db = FUSE ? (dinv + rbase) : nullptr;
        float*       ab = FUSE ? (AGG + (size_t)rbase * DHID + col) : nullptr;

        if (r0 + 16 <= M) {            // fast path: whole subtile valid (M%16==0 here)
            #pragma unroll
            for (int p = 0; p < 8; ++p) {
                cb[p * DHID] = acc[p];
                if (FUSE) {
                    float di = db[p];
                    ab[p * DHID] = acc[p] * di * di;
                }
            }
        } else {                        // tail fallback (uniform branch)
            #pragma unroll
            for (int p = 0; p < 8; ++p) {
                if (rbase + p < M) {
                    cb[p * DHID] = acc[p];
                    if (FUSE) {
                        float di = db[p];
                        ab[p * DHID] = acc[p] * di * di;
                    }
                }
            }
        }
    }
}

// --------------------------------------------------- GCN aggregation kernels
// one wave per edge; lane covers 4 channels
__global__ void k_edge_scatter(const int* __restrict__ src, const int* __restrict__ dst,
                               const float* __restrict__ dinv, const float* __restrict__ H,
                               float* __restrict__ AGG, int E) {
    int gid  = blockIdx.x * blockDim.x + threadIdx.x;
    int e    = gid >> 5;
    int lane = gid & 31;
    if (e >= E) return;
    int s = src[e];
    int d = dst[e];
    float nrm = dinv[s] * dinv[d];
    float4 v = ((const float4*)(H + (size_t)s * DHID))[lane];
    float* out = AGG + (size_t)d * DHID + lane * 4;
    atomicAdd(out + 0, v.x * nrm);
    atomicAdd(out + 1, v.y * nrm);
    atomicAdd(out + 2, v.z * nrm);
    atomicAdd(out + 3, v.w * nrm);
}

// out16 = f16(relu(AGG + b)), 4 channels / thread
__global__ void k_bias_relu_f16(const float* __restrict__ AGG, const float* __restrict__ b,
                                _Float16* __restrict__ out16, int n32) {
    int idx = blockIdx.x * blockDim.x + threadIdx.x;   // n * 32
    if (idx >= n32) return;
    int c4 = idx & 31;
    float4 v  = ((const float4*)AGG)[idx];
    float4 bv = ((const float4*)b)[c4];
    v4h h = { (_Float16)fmaxf(v.x + bv.x, 0.0f),
              (_Float16)fmaxf(v.y + bv.y, 0.0f),
              (_Float16)fmaxf(v.z + bv.z, 0.0f),
              (_Float16)fmaxf(v.w + bv.w, 0.0f) };
    ((v4h*)out16)[idx] = h;
}

// ---------------------------------------------------------------- pooling
__global__ void k_pool_count(const int* __restrict__ batch, float* __restrict__ cnt, int n) {
    int i = blockIdx.x * blockDim.x + threadIdx.x;
    if (i < n) atomicAdd(&cnt[batch[i]], 1.0f);
}

__global__ void k_pool_sum(const float* __restrict__ A, const int* __restrict__ batch,
                           float* __restrict__ sums, int n) {
    int idx = blockIdx.x * blockDim.x + threadIdx.x;   // n * 32
    if (idx >= n * 32) return;
    int node = idx >> 5;
    int c4   = idx & 31;
    int g    = batch[node];
    float4 v = ((const float4*)(A + (size_t)node * DHID))[c4];
    float* out = sums + (size_t)g * DHID + c4 * 4;
    atomicAdd(out + 0, v.x);
    atomicAdd(out + 1, v.y);
    atomicAdd(out + 2, v.z);
    atomicAdd(out + 3, v.w);
}

// ge = sums/cnt + bfc1 -> d_out[0..65535]; penalty -> d_out[65536]
__global__ void k_mean_bias(const float* __restrict__ sums, const float* __restrict__ cnt,
                            const float* __restrict__ bfc1, float* __restrict__ out) {
    int idx = blockIdx.x * blockDim.x + threadIdx.x;   // 512*128
    if (idx >= NGRAPH * DHID) return;
    int g = idx >> 7;
    int c = idx & (DHID - 1);
    float cc = fmaxf(cnt[g], 1.0f);
    out[idx] = sums[idx] / cc + bfc1[c];
    if (idx == 0) out[NGRAPH * DHID] = 0.0f;           // penalty
}

// ic = ge @ Wfc2 + bfc2 (tiny: 512x10, K=128)
__global__ void k_head(const float* __restrict__ ge, const float* __restrict__ Wfc2,
                       const float* __restrict__ bfc2, float* __restrict__ out) {
    int idx = blockIdx.x * blockDim.x + threadIdx.x;   // 5120
    if (idx >= NGRAPH * 10) return;
    int g = idx / 10;
    int o = idx % 10;
    float acc = bfc2[o];
    const float* row = ge + (size_t)g * DHID;
    #pragma unroll 8
    for (int c = 0; c < DHID; ++c) acc += row[c] * Wfc2[c * 10 + o];
    out[idx] = acc;
}

// ---------------------------------------------------------------- launcher
static inline size_t align_up(size_t x, size_t a) { return (x + a - 1) & ~(a - 1); }

extern "C" void kernel_launch(void* const* d_in, const int* in_sizes, int n_in,
                              void* d_out, int out_size, void* d_ws, size_t ws_size,
                              hipStream_t stream) {
    const float* x     = (const float*)d_in[0];
    const int*   ei    = (const int*)  d_in[1];
    const int*   batch = (const int*)  d_in[2];
    const float* W1    = (const float*)d_in[3];
    const float* b1    = (const float*)d_in[4];
    const float* W2    = (const float*)d_in[5];
    const float* b2    = (const float*)d_in[6];
    const float* Wfc1  = (const float*)d_in[7];
    const float* bfc1  = (const float*)d_in[8];
    const float* Wfc2  = (const float*)d_in[9];
    const float* bfc2  = (const float*)d_in[10];

    const int n = in_sizes[0] / DHID;       // 100000
    const int E = in_sizes[1] / 2;          // 1600000
    const int* src = ei;
    const int* dst = ei + E;

    // workspace carve-out
    char* w = (char*)d_ws;
    size_t off = 0;
    float*    dinv = (float*)(w + off);    off = align_up(off + (size_t)n * 4, 256);
    _Float16* in16 = (_Float16*)(w + off); off = align_up(off + (size_t)n * DHID * 2, 256);
    float*    H    = (float*)(w + off);    off = align_up(off + (size_t)n * DHID * 4, 256);
    float*    AGG  = (float*)(w + off);    off = align_up(off + (size_t)n * DHID * 4, 256);
    _Float16* Wp   = (_Float16*)(w + off); off = align_up(off + (size_t)16384 * 2, 256);
    float*    sums = (float*)(w + off);    off = align_up(off + (size_t)NGRAPH * DHID * 4, 256);
    float*    cnt  = (float*)(w + off);    off = align_up(off + (size_t)NGRAPH * 4, 256);
    (void)ws_size;

    float* out = (float*)d_out;
    const int B = 256;
    auto g1 = [&](long total) { return dim3((unsigned)((total + B - 1) / B)); };

    // ---- degree / dinv ----
    k_fill_f32<<<g1(n), B, 0, stream>>>(dinv, 1.0f, n);
    k_deg<<<g1(E), B, 0, stream>>>(dst, dinv, E);
    k_rsqrt_inplace<<<g1(n), B, 0, stream>>>(dinv, n);

    // ---- features -> f16 (4 ch / thread) ----
    k_f32_to_f16_v4<<<g1((long)n * 32), B, 0, stream>>>(x, in16, n * 32);

    const int gemmBlocks = (n + 63) / 64;

    // ---- layer 1: H = in16 @ W1 (AGG = H*dinv^2 fused) ; scatter ; relu ----
    k_repack_w<<<g1(16384), B, 0, stream>>>(W1, Wp);
    k_gemm_wmma<true><<<gemmBlocks, B, 0, stream>>>(in16, Wp, H, dinv, AGG, n);
    k_edge_scatter<<<g1((long)E * 32), B, 0, stream>>>(src, dst, dinv, H, AGG, E);
    k_bias_relu_f16<<<g1((long)n * 32), B, 0, stream>>>(AGG, b1, in16, n * 32);

    // ---- layer 2 ----
    k_repack_w<<<g1(16384), B, 0, stream>>>(W2, Wp);
    k_gemm_wmma<true><<<gemmBlocks, B, 0, stream>>>(in16, Wp, H, dinv, AGG, n);
    k_edge_scatter<<<g1((long)E * 32), B, 0, stream>>>(src, dst, dinv, H, AGG, E);
    k_bias_relu_f16<<<g1((long)n * 32), B, 0, stream>>>(AGG, b2, in16, n * 32);

    // ---- fc1 (no relu; bias folded into mean): H = in16 @ Wfc1 ----
    k_repack_w<<<g1(16384), B, 0, stream>>>(Wfc1, Wp);
    k_gemm_wmma<false><<<gemmBlocks, B, 0, stream>>>(in16, Wp, H, nullptr, nullptr, n);

    // ---- global mean pool ----
    k_zero_f32<<<g1(NGRAPH * DHID + NGRAPH), B, 0, stream>>>(sums, NGRAPH * DHID + NGRAPH); // cnt contiguous after sums
    k_pool_count<<<g1(n), B, 0, stream>>>(batch, cnt, n);
    k_pool_sum<<<g1((long)n * 32), B, 0, stream>>>(H, batch, sums, n);
    k_mean_bias<<<g1(NGRAPH * DHID), B, 0, stream>>>(sums, cnt, bfc1, out);

    // ---- head: ic = ge @ Wfc2 + bfc2 ----
    k_head<<<g1(NGRAPH * 10), B, 0, stream>>>(out, Wfc2, bfc2, out + NGRAPH * DHID + 1);
}
